// Atten_996432413050
// MI455X (gfx1250) — matrix operand order repaired and checked by reading, not compiled, not run on hardware
//
#include <hip/hip_runtime.h>
#include <math.h>

#define B_ 64
#define R_ 49
#define T_ 512
#define H_ 512
#define K_ 49

typedef __attribute__((ext_vector_type(16))) __bf16 v16bf;
typedef __attribute__((ext_vector_type(8)))  float  v8f;

// ---------------- WMMA operand loaders (v_wmma_f32_16x16x32_bf16) -------------
// A operand: 16(M) x 32(K). Lane L holds row m = mbase + (L%16);
// v[0..7]  = K (L/16)*8 .. +7 ;  v[8..15] = K 16+(L/16)*8 .. +7.
__device__ __forceinline__ v16bf load_a_f32(const float* __restrict__ base,
                                            int stride, int mbase, int kbase, int lane) {
  int m  = mbase + (lane & 15);
  int c0 = kbase + ((lane >> 4) << 3);
  const float* p = base + (size_t)m * stride + c0;
  v16bf r;
#pragma unroll
  for (int i = 0; i < 8; ++i) {
    r[i]     = (__bf16)p[i];
    r[i + 8] = (__bf16)p[i + 16];
  }
  return r;
}

__device__ __forceinline__ v16bf load_a_bf16(const __bf16* base,
                                             int stride, int mbase, int kbase, int lane) {
  int m  = mbase + (lane & 15);
  int c0 = kbase + ((lane >> 4) << 3);
  const __bf16* p = base + (size_t)m * stride + c0;
  v16bf r;
#pragma unroll
  for (int i = 0; i < 8; ++i) {
    r[i]     = p[i];
    r[i + 8] = p[i + 16];
  }
  return r;
}

// B operand: 32(K) x 16(N) from N-major storage (N rows, K contiguous cols).
// Lane L holds col n = nbase + (L%16); v[0..7] = K (L/16)*8 .. +7 ; v[8..15] = K 16+(L/16)*8 .. +7.
__device__ __forceinline__ v16bf load_b_bf16(const __bf16* __restrict__ base,
                                             int stride, int nbase, int kbase, int lane) {
  int n  = nbase + (lane & 15);
  int c0 = kbase + ((lane >> 4) << 3);
  typedef __attribute__((ext_vector_type(8))) __bf16 v8bf;
  const __bf16* p = base + (size_t)n * stride + c0;
  v8bf lo = *(const v8bf*)(p);
  v8bf hi = *(const v8bf*)(p + 16);
  return __builtin_shufflevector(lo, hi, 0,1,2,3,4,5,6,7,8,9,10,11,12,13,14,15);
}

// ---------------- K0: convert/pad weights, build V^T in bf16 -----------------
__global__ __launch_bounds__(256) void atten_prep(
    const float* __restrict__ V, const float* __restrict__ Wv,
    const float* __restrict__ Wg, const float* __restrict__ Ws,
    const float* __restrict__ Wh,
    __bf16* __restrict__ VTbf, __bf16* __restrict__ Wvbf,
    __bf16* __restrict__ Wgbf, __bf16* __restrict__ Wsbf,
    float* __restrict__ whp) {
  int b = blockIdx.x;
  if (b < B_) {
    // VTbf[b][h][r] = V[b][r][h], r padded to 64 with zeros
    for (int idx = threadIdx.x; idx < H_ * 64; idx += 256) {
      int h = idx >> 6, r = idx & 63;
      float v = (r < R_) ? V[((size_t)b * R_ + r) * H_ + h] : 0.f;
      VTbf[(size_t)b * H_ * 64 + idx] = (__bf16)v;
    }
  } else {
    for (int idx = threadIdx.x; idx < 64 * H_; idx += 256) {
      int k = idx >> 9, h = idx & (H_ - 1);
      float wv = (k < K_) ? Wv[(size_t)k * H_ + h] : 0.f;
      float wg = (k < K_) ? Wg[(size_t)k * H_ + h] : 0.f;
      float ws = (k < K_) ? Ws[(size_t)k * H_ + h] : 0.f;
      Wvbf[idx] = (__bf16)wv;
      Wgbf[idx] = (__bf16)wg;
      Wsbf[idx] = (__bf16)ws;
    }
    if (threadIdx.x < 64) whp[threadIdx.x] = (threadIdx.x < K_) ? Wh[threadIdx.x] : 0.f;
  }
}

// ---------------- K1: cv[b,r,k] = sum_h V[b,r,h] * Wv[k,h]  (f32, tiny) ------
__global__ __launch_bounds__(256) void atten_cv(
    const float* __restrict__ V, const float* __restrict__ Wv,
    float* __restrict__ cv) {
  int b = blockIdx.x;
  for (int idx = threadIdx.x; idx < 64 * 64; idx += 256) {
    int r = idx >> 6, k = idx & 63;
    float acc = 0.f;
    if (r < R_ && k < K_) {
      const float* vr = V + ((size_t)b * R_ + r) * H_;
      const float* wr = Wv + (size_t)k * H_;
#pragma unroll 4
      for (int h = 0; h < H_; ++h) acc = fmaf(vr[h], wr[h], acc);
    }
    cv[((size_t)b * 64 + r) * 64 + k] = acc;
  }
}

// ---------------- K2: gh = h_t*Wg^T, sv = s_t*Ws^T, z_ext fused (WMMA) -------
__global__ __launch_bounds__(256) void atten_gh_zext(
    const float* __restrict__ h_t, const float* __restrict__ s_t,
    const __bf16* __restrict__ Wgbf, const __bf16* __restrict__ Wsbf,
    const float* __restrict__ whp,
    float* __restrict__ gh_out, float* __restrict__ zext_out) {
  int wave = threadIdx.x >> 5;
  int lane = threadIdx.x & 31;
  int b = blockIdx.x >> 2;
  int tbase = (blockIdx.x & 3) * 128 + wave * 16;   // 16-t strip per wave

  const float* hrow = h_t + ((size_t)b * T_ + tbase) * H_;
  const float* srow = s_t + ((size_t)b * T_ + tbase) * H_;

  v8f gC[4] = {};
  v8f sC[4] = {};
  for (int ks = 0; ks < H_; ks += 32) {
    v16bf ah = load_a_f32(hrow, H_, 0, ks, lane);
    v16bf as = load_a_f32(srow, H_, 0, ks, lane);
#pragma unroll
    for (int j = 0; j < 4; ++j) {
      v16bf bg = load_b_bf16(Wgbf, H_, j * 16, ks, lane);
      v16bf bs = load_b_bf16(Wsbf, H_, j * 16, ks, lane);
      gC[j] = __builtin_amdgcn_wmma_f32_16x16x32_bf16(false, ah, false, bg,
                                                      (short)0, gC[j], false, false);
      sC[j] = __builtin_amdgcn_wmma_f32_16x16x32_bf16(false, as, false, bs,
                                                      (short)0, sC[j], false, false);
    }
  }

  // D layout: value (m, n) with m = (lane/16)*8 + i, n = j*16 + (lane%15... %16)
  int half = lane >> 4;
  int nlo  = lane & 15;
  float wl[4];
#pragma unroll
  for (int j = 0; j < 4; ++j) wl[j] = whp[j * 16 + nlo];

  float* ghrow = gh_out + ((size_t)b * T_ + tbase) * 64;
  float zp[8];
#pragma unroll
  for (int i = 0; i < 8; ++i) zp[i] = 0.f;
#pragma unroll
  for (int j = 0; j < 4; ++j) {
#pragma unroll
    for (int i = 0; i < 8; ++i) {
      int m = half * 8 + i;
      int n = j * 16 + nlo;
      float g = gC[j][i];
      ghrow[(size_t)m * 64 + n] = g;
      zp[i] += tanhf(g + sC[j][i]) * wl[j];
    }
  }
  // reduce over the 16 lanes sharing each m (xor within half-wave)
#pragma unroll
  for (int d = 1; d < 16; d <<= 1) {
#pragma unroll
    for (int i = 0; i < 8; ++i) zp[i] += __shfl_xor(zp[i], d, 32);
  }
  if (nlo == 0) {
#pragma unroll
    for (int i = 0; i < 8; ++i)
      zext_out[(size_t)b * T_ + tbase + half * 8 + i] = zp[i];
  }
}

// ---------------- K3: z_t/tanh, softmax, beta, c_t GEMM (WMMA), blend --------
__global__ __launch_bounds__(256) void atten_main(
    const float* __restrict__ s_t, const __bf16* __restrict__ VTbf,
    const float* __restrict__ cv, const float* __restrict__ gh,
    const float* __restrict__ zext, const float* __restrict__ whp,
    float* __restrict__ chat_out, float* __restrict__ alpha_out,
    float* __restrict__ beta_out) {
  __shared__ float cv_s[R_ * 65];                 // stride 65 -> conflict-free
  __shared__ float gh_s[64 * 64];
  __shared__ float wh_s[64];
  __shared__ float beta_s[64];
  __shared__ __align__(16) __bf16 al_s[64 * 72];  // stride 72 -> conflict-free A tile

  int b = blockIdx.x >> 3;
  int tbase = (blockIdx.x & 7) * 64;
  int tid = threadIdx.x;
  int lane = tid & 31;
  int wave = tid >> 5;

  for (int idx = tid; idx < R_ * 64; idx += 256) {
    int r = idx >> 6, k = idx & 63;
    cv_s[r * 65 + k] = cv[((size_t)b * 64 + r) * 64 + k];
  }
  for (int idx = tid; idx < 64 * 64; idx += 256)
    gh_s[idx] = gh[((size_t)b * T_ + tbase) * 64 + idx];
  if (tid < 64) wh_s[tid] = whp[tid];
  __syncthreads();

  // ---- phase 2: z_t, softmax over R, extended softmax -> beta ----
  int r1 = lane;
  int r2 = lane + 32;
  bool v2 = (r2 < R_);
  const float* cv1 = cv_s + r1 * 65;
  const float* cv2 = cv_s + (v2 ? r2 : 0) * 65;
  for (int ti = 0; ti < 8; ++ti) {
    int tl = wave * 8 + ti;
    int t = tbase + tl;
    const float* ghr = gh_s + tl * 64;
    float z1 = 0.f, z2 = 0.f;
    for (int k = 0; k < K_; ++k) {
      float g = ghr[k];
      float w = wh_s[k];
      z1 += tanhf(cv1[k] + g) * w;
      z2 += tanhf(cv2[k] + g) * w;
    }
    float m = fmaxf(z1, v2 ? z2 : -INFINITY);
#pragma unroll
    for (int d = 1; d < 32; d <<= 1) m = fmaxf(m, __shfl_xor(m, d, 32));
    float e1 = __expf(z1 - m);
    float e2 = v2 ? __expf(z2 - m) : 0.f;
    float s = e1 + e2;
#pragma unroll
    for (int d = 1; d < 32; d <<= 1) s += __shfl_xor(s, d, 32);
    float inv = 1.f / s;
    float a1 = e1 * inv;
    float a2 = e2 * inv;
    size_t abase = ((size_t)b * T_ + t) * K_;
    alpha_out[abase + r1] = a1;
    if (v2) alpha_out[abase + r2] = a2;
    al_s[tl * 72 + r1] = (__bf16)a1;
    al_s[tl * 72 + r2] = (__bf16)(v2 ? a2 : 0.f);   // also zeroes pad rows 49..63
    // extended softmax (shares exps with the R-softmax)
    float ze = zext[(size_t)b * T_ + t];
    float m2 = fmaxf(m, ze);
    float s2 = s * __expf(m - m2) + __expf(ze - m2);
    float beta = __expf(ze - m2) / s2;
    if (lane == 0) {
      beta_s[tl] = beta;
      beta_out[(size_t)b * T_ + t] = beta;
    }
  }
  __syncthreads();

  // ---- phase 3: c_t = alpha * V via WMMA, fused beta-blend with s_t ----
  int tstrip = wave >> 1;              // 4 strips of 16 t
  int hbase0 = (wave & 1) * 256;       // each strip split across 2 waves in H
  const __bf16* vt = VTbf + (size_t)b * H_ * 64;

  v16bf a0 = load_a_bf16(al_s, 72, tstrip * 16, 0, lane);
  v16bf a1 = load_a_bf16(al_s, 72, tstrip * 16, 32, lane);
  int half = lane >> 4;
  int nlo = lane & 15;
  for (int nt = 0; nt < 16; ++nt) {
    int hb = hbase0 + nt * 16;
    v16bf b0 = load_b_bf16(vt, 64, hb, 0, lane);
    v16bf b1 = load_b_bf16(vt, 64, hb, 32, lane);
    v8f c = {};
    c = __builtin_amdgcn_wmma_f32_16x16x32_bf16(false, a0, false, b0,
                                                (short)0, c, false, false);
    c = __builtin_amdgcn_wmma_f32_16x16x32_bf16(false, a1, false, b1,
                                                (short)0, c, false, false);
#pragma unroll
    for (int i = 0; i < 8; ++i) {
      int m = half * 8 + i;
      int t = tbase + tstrip * 16 + m;
      int h = hb + nlo;
      size_t idx = ((size_t)b * T_ + t) * H_ + h;
      float sv = s_t[idx];
      float bt = beta_s[tstrip * 16 + m];
      chat_out[idx] = bt * sv + (1.f - bt) * c[i];
    }
  }
}

// -----------------------------------------------------------------------------
extern "C" void kernel_launch(void* const* d_in, const int* in_sizes, int n_in,
                              void* d_out, int out_size, void* d_ws, size_t ws_size,
                              hipStream_t stream) {
  const float* V   = (const float*)d_in[0];
  const float* h_t = (const float*)d_in[1];
  const float* s_t = (const float*)d_in[2];
  const float* Wv  = (const float*)d_in[3];
  const float* Wg  = (const float*)d_in[4];
  const float* Ws  = (const float*)d_in[5];
  const float* Wh  = (const float*)d_in[6];

  float* out   = (float*)d_out;
  float* chat  = out;                                   // B*T*H
  float* alpha = chat + (size_t)B_ * T_ * H_;           // B*T*R
  float* beta  = alpha + (size_t)B_ * T_ * R_;          // B*T

  char* ws = (char*)d_ws;
  size_t off = 0;
  auto carve = [&](size_t bytes) -> char* {
    char* p = ws + off;
    off = (off + bytes + 255) & ~(size_t)255;
    return p;
  };
  __bf16* VTbf = (__bf16*)carve((size_t)B_ * H_ * 64 * 2);   // 4 MB
  __bf16* Wvbf = (__bf16*)carve((size_t)64 * H_ * 2);
  __bf16* Wgbf = (__bf16*)carve((size_t)64 * H_ * 2);
  __bf16* Wsbf = (__bf16*)carve((size_t)64 * H_ * 2);
  float*  whp  = (float*)carve(64 * 4);
  float*  cvw  = (float*)carve((size_t)B_ * 64 * 64 * 4);    // 1 MB
  float*  ghw  = (float*)carve((size_t)B_ * T_ * 64 * 4);    // 8 MB
  float*  zext = (float*)carve((size_t)B_ * T_ * 4);
  (void)ws_size; (void)in_sizes; (void)n_in; (void)out_size;

  atten_prep<<<B_ + 1, 256, 0, stream>>>(V, Wv, Wg, Ws, Wh, VTbf, Wvbf, Wgbf, Wsbf, whp);
  atten_cv<<<B_, 256, 0, stream>>>(V, Wv, cvw);
  atten_gh_zext<<<B_ * 4, 256, 0, stream>>>(h_t, s_t, Wgbf, Wsbf, whp, ghw, zext);
  atten_main<<<B_ * 8, 256, 0, stream>>>(s_t, VTbf, cvw, ghw, zext, whp, chat, alpha, beta);
}
